// GraphMamba_22497038696553
// MI455X (gfx1250) — matrix operand order repaired and checked by reading, not compiled, not run on hardware
//
#include <hip/hip_runtime.h>
#include <hip/hip_bf16.h>

// ---------------------------------------------------------------------------
// GraphMamba block for MI455X (gfx1250, wave32, WMMA + Tensor Data Mover).
//
// The 3-branch (vertex-mix -> channel-mix) operator per head collapses into a
// single 192x192 matrix M[h] on the packed (c,v) feature:
//     y[n,h,t,:] = M[h] @ x[n,h,t,:] + bias[h]
// Pass plan (recompute to avoid a 189 MB intermediate):
//   stage0   : build M[h] in f16, pre-swizzled into WMMA B-fragment layout
//   memset   : zero BN accumulators
//   gm_main 0: GEMM (TDM-fed, double buffered), per-channel sum/sumsq
//   finalize : per-channel scale/shift
//   gm_main 1: GEMM again, BN + residual (from LDS) + relu, store
// ---------------------------------------------------------------------------

typedef __attribute__((ext_vector_type(16))) _Float16 v16h;
typedef __attribute__((ext_vector_type(8)))  float    v8f;
typedef __attribute__((ext_vector_type(4)))  unsigned int u32x4;
typedef __attribute__((ext_vector_type(8)))  unsigned int u32x8;

#define GV   24                 // vertices
#define GH   8                  // heads
#define GCG  8                  // channels per head
#define NF   192                // GCG*GV combined feature
#define TOK  48                 // tokens (t values) per chunk
#define NCHUNK 5                // 240 / TOK
#define KT_N 6                  // K tiles   (192/32)
#define NT_N 12                 // out tiles (192/16)
#define FRAG 512                // halfs per B fragment (32 lanes * 16)
#define WPH  (KT_N*NT_N*FRAG)   // 36864 halfs per head

// LDS layout (bytes): B pack | x buf0 | x buf1 | stats
#define LDS_B   0
#define LDS_X0  73728
#define LDS_X1  110592          // + 48*192*4
#define LDS_ST  147456
#define LDS_TOT (147456 + 512)

// ---------------------------------------------------------------------------
// TDM: issue a 2D tensor_load_to_lds. D# bitfields per CDNA5 ISA.
// 2-group form (tensors up to 2D): group0 = 4 SGPRs, group1 = 8 SGPRs.
// ---------------------------------------------------------------------------
__device__ __forceinline__ void tdm_load_2d(unsigned lds_addr, const void* gaddr,
                                            unsigned tensor_d0, unsigned tensor_d1,
                                            unsigned tile_d0,   unsigned tile_d1,
                                            unsigned long long d0_stride,
                                            unsigned data_size_log) // 0:1B 1:2B 2:4B 3:8B
{
    unsigned long long ga = (unsigned long long)(size_t)gaddr;
    u32x4 g0;
    g0[0] = 1u;                                               // count=1, user D#
    g0[1] = lds_addr;                                         // lds_addr [63:32]
    g0[2] = (unsigned)(ga & 0xffffffffu);                     // global_addr[31:0]
    g0[3] = (unsigned)((ga >> 32) & 0x01ffffffu) | 0x80000000u; // addr[56:32] | type=2
    u32x8 g1;
    g1[0] = data_size_log << 16;                              // wg_mask=0 | data_size
    g1[1] = (tensor_d0 & 0xffffu) << 16;                      // tensor_dim0[15:0] @ bit48
    g1[2] = ((tensor_d0 >> 16) & 0xffffu) | ((tensor_d1 & 0xffffu) << 16);
    g1[3] = ((tensor_d1 >> 16) & 0xffffu) | ((tile_d0 & 0xffffu) << 16);
    g1[4] = (tile_d1 & 0xffffu);                              // tile_dim1 | tile_dim2=0
    g1[5] = (unsigned)(d0_stride & 0xffffffffu);              // dim0_stride[31:0]
    g1[6] = (unsigned)((d0_stride >> 32) & 0xffffu);          // dim0_stride[47:32] | dim1_stride.lo=0
    g1[7] = 0u;
    asm volatile("tensor_load_to_lds %0, %1" :: "s"(g0), "s"(g1) : "memory");
}

__device__ __forceinline__ unsigned lds_off(const void* p)
{   // generic LDS pointer: low 32 bits are the wave-relative LDS byte address
    return (unsigned)(size_t)p;
}

// ---------------------------------------------------------------------------
// Stage 0: per-head combined operator, pre-swizzled for the WMMA B layout.
//   W[cv][ow] = sum_i w2[i,h,o,c] * w1_i[h,v,w]
//   w1_i = fc1_i/l2n(fc1_i) + pos_i/l2n(pos_i), pos_i[v,w] = rpe[i,h,hops[v,w]]
// B-frag element (lane,e): K = e + 16*(lane>>4), N = lane&15   (contiguous k order, same as A)
// ---------------------------------------------------------------------------
__global__ void gm_stage0(const float* __restrict__ fc1, const float* __restrict__ rpe,
                          const float* __restrict__ w2,  const float* __restrict__ b2,
                          const int* __restrict__ hops,
                          _Float16* __restrict__ wpack, float* __restrict__ bsum)
{
    __shared__ float w1sh[3][GV][GV];
    __shared__ float invF[3][GV], invP[3][GV];
    const int tid = threadIdx.x;

    for (int h = 0; h < GH; ++h) {
        if (tid < 3 * GV) {
            int i = tid / GV, w = tid % GV;
            float sF = 0.f, sP = 0.f;
            for (int v = 0; v < GV; ++v) {
                float f = fc1[(((i * GH + h) * GV) + v) * GV + w];
                float p = rpe[(i * GH + h) * GV + hops[v * GV + w]];
                sF += f * f;  sP += p * p;
            }
            invF[i][w] = 1.0f / (sqrtf(sF) + 1e-4f);
            invP[i][w] = 1.0f / (sqrtf(sP) + 1e-4f);
        }
        __syncthreads();
        for (int idx = tid; idx < 3 * GV * GV; idx += blockDim.x) {
            int i = idx / (GV * GV), r = idx % (GV * GV);
            int v = r / GV, w = r % GV;
            float f = fc1[(((i * GH + h) * GV) + v) * GV + w];
            float p = rpe[(i * GH + h) * GV + hops[v * GV + w]];
            w1sh[i][v][w] = f * invF[i][w] + p * invP[i][w];
        }
        if (tid < GCG) {
            int ch = h * GCG + tid;
            bsum[ch] = b2[ch] + b2[64 + ch] + b2[128 + ch];
        }
        __syncthreads();

        _Float16* wp = wpack + (size_t)h * WPH;
        for (int j = tid; j < WPH; j += blockDim.x) {
            int e    = j & 15;
            int lane = (j >> 4) & 31;
            int nt   = (j >> 9) % NT_N;
            int kt   = (j >> 9) / NT_N;
            int cv   = kt * 32 + e + 16 * (lane >> 4);
            int ow   = nt * 16 + (lane & 15);
            int c = cv / GV, v = cv % GV;
            int o = ow / GV, w = ow % GV;
            float val = 0.f;
            #pragma unroll
            for (int i = 0; i < 3; ++i)
                val += w2[((i * GH + h) * GCG + o) * GCG + c] * w1sh[i][v][w];
            wp[j] = (_Float16)val;
        }
        __syncthreads();
    }
}

// ---------------------------------------------------------------------------
// Main GEMM: persistent block per (head, n). 96 threads = 3 waves, each wave
// owns a 16-token row tile across all 192 output columns. TDM streams the
// 48-token x chunks into double-buffered LDS while WMMA crunches the other.
// mode 0: BN stat accumulation.   mode 1: BN + residual + relu + store.
// ---------------------------------------------------------------------------
__global__ void __launch_bounds__(96) gm_main(
    const float* __restrict__ x, const _Float16* __restrict__ wpack,
    const float* __restrict__ bsum,
    float* __restrict__ gsum, float* __restrict__ gsumsq,
    const float* __restrict__ scaleA, const float* __restrict__ shiftA,
    float* __restrict__ out, int mode)
{
    extern __shared__ char smem[];
    _Float16* bsm    = (_Float16*)(smem + LDS_B);   // head operator, frag layout
    float*    xsb0   = (float*)(smem + LDS_X0);     // x chunk: [c][t][v] tight, 9216 f32
    float*    xsb1   = (float*)(smem + LDS_X1);
    float*    statsL = (float*)(smem + LDS_ST);     // 128 f32

    const int tid = threadIdx.x;
    const int h   = blockIdx.x;
    const int n   = blockIdx.y;
    const int lane  = tid & 31;
    const int wv    = tid >> 5;
    const int m0    = wv * 16;
    const int row   = m0 + (lane & 15);
    const int khalf = lane >> 4;

    const float* xg0 = x + (((size_t)(n * 64 + h * 8)) * 240) * 24;  // c=0,t=0

    // Prologue: TDM-load B pack (1 row x 73728 B) and x chunk 0 (8 x 1152 f32).
    if (wv == 0) {
        tdm_load_2d(lds_off(bsm), wpack + (size_t)h * WPH,
                    WPH, 1, WPH, 1, WPH, 1u /*2B*/);
        tdm_load_2d(lds_off(xsb0), xg0,
                    TOK * GV, GCG, TOK * GV, GCG, 240u * GV, 2u /*4B*/);
    }
    if (mode == 0) { for (int j = tid; j < 128; j += 96) statsL[j] = 0.f; }
    if (wv == 0) __builtin_amdgcn_s_wait_tensorcnt(0);
    __syncthreads();

    for (int tc = 0; tc < NCHUNK; ++tc) {
        float* xs = (tc & 1) ? xsb1 : xsb0;
        // Prefetch next chunk into the other buffer while we compute.
        if (wv == 0 && tc + 1 < NCHUNK) {
            float* xn = (tc & 1) ? xsb0 : xsb1;
            tdm_load_2d(lds_off(xn), xg0 + (size_t)(tc + 1) * TOK * GV,
                        TOK * GV, GCG, TOK * GV, GCG, 240u * GV, 2u);
        }

        // Build 6 A fragments (K=192) in contiguous k order:
        // VGPR p: K=2p,2p+1 ; lanes 16-31 add 16 (matches B pack K order).
        v16h afr[KT_N];
        #pragma unroll
        for (int kt = 0; kt < KT_N; ++kt) {
            #pragma unroll
            for (int p = 0; p < 8; ++p) {
                int K  = 2 * p + 16 * khalf;
                int cv = kt * 32 + K;
                int c  = cv / GV, v = cv - c * GV;        // pair never crosses c (cv even)
                const float2 f = *(const float2*)&xs[c * (TOK * GV) + row * GV + v];
                afr[kt][2 * p]     = (_Float16)f.x;
                afr[kt][2 * p + 1] = (_Float16)f.y;
            }
        }

        for (int nt = 0; nt < NT_N; ++nt) {
            v8f acc = {};
            #pragma unroll
            for (int kt = 0; kt < KT_N; ++kt) {
                const v16h b = *(const v16h*)&bsm[((kt * NT_N + nt) * 32 + lane) * 16];
                acc = __builtin_amdgcn_wmma_f32_16x16x32_f16(
                          false, afr[kt], false, b, (short)0, acc, false, false);
            }
            // D layout: VGPR r -> row m0+r (lanes<16) / m0+r+8 (lanes>=16); col = lane&15
            const int ow = nt * 16 + (lane & 15);
            const int o  = ow / GV, w = ow - o * GV;
            const int ch = h * GCG + o;
            const float bias = bsum[ch];
            if (mode == 0) {
                float s = 0.f, s2 = 0.f;
                #pragma unroll
                for (int r = 0; r < 8; ++r) {
                    float yv = acc[r] + bias;
                    s += yv;  s2 += yv * yv;
                }
                atomicAdd(&statsL[ch], s);
                atomicAdd(&statsL[64 + ch], s2);
            } else {
                const float sc = scaleA[ch], sh = shiftA[ch];
                #pragma unroll
                for (int r = 0; r < 8; ++r) {
                    int   tr  = m0 + r + 8 * khalf;
                    float yv  = acc[r] + bias;
                    float res = xs[o * (TOK * GV) + tr * GV + w];  // residual from LDS
                    float val = sc * yv + sh + res;
                    out[(((size_t)(n * 64 + ch)) * 240 + (tc * TOK + tr)) * GV + w] =
                        fmaxf(val, 0.f);
                }
            }
        }

        if (wv == 0) __builtin_amdgcn_s_wait_tensorcnt(0);  // next chunk landed
        __syncthreads();                                     // + this chunk's reads done
    }

    if (mode == 0) {
        for (int j = tid; j < 64; j += 96) {
            atomicAdd(&gsum[j],   statsL[j]);
            atomicAdd(&gsumsq[j], statsL[64 + j]);
        }
    }
}

// ---------------------------------------------------------------------------
__global__ void gm_finalize(const float* __restrict__ gsum, const float* __restrict__ gsumsq,
                            const float* __restrict__ gamma, const float* __restrict__ beta,
                            float* __restrict__ scaleA, float* __restrict__ shiftA)
{
    int c = threadIdx.x;
    if (c < 64) {
        const float cnt = 128.0f * 240.0f * 24.0f;
        float mean = gsum[c] / cnt;
        float var  = gsumsq[c] / cnt - mean * mean;
        float sc   = gamma[c] * rsqrtf(var + 1e-5f);
        scaleA[c]  = sc;
        shiftA[c]  = beta[c] - mean * sc;
    }
}

// ---------------------------------------------------------------------------
extern "C" void kernel_launch(void* const* d_in, const int* in_sizes, int n_in,
                              void* d_out, int out_size, void* d_ws, size_t ws_size,
                              hipStream_t stream)
{
    const float* x     = (const float*)d_in[0];
    const float* fc1   = (const float*)d_in[1];
    const float* rpe   = (const float*)d_in[2];
    const float* w2    = (const float*)d_in[3];
    const float* b2    = (const float*)d_in[4];
    const float* gamma = (const float*)d_in[5];
    const float* beta  = (const float*)d_in[6];
    const int*   hops  = (const int*)d_in[7];
    float* out = (float*)d_out;

    char* ws = (char*)d_ws;                       // total use < 1 MB
    _Float16* wpack  = (_Float16*)ws;             // 8 * 36864 halfs = 589824 B
    float*    bsum   = (float*)(ws + 589824);
    float*    gsum   = (float*)(ws + 589824 + 256);
    float*    gsumsq = (float*)(ws + 589824 + 512);
    float*    scaleA = (float*)(ws + 589824 + 768);
    float*    shiftA = (float*)(ws + 589824 + 1024);

    gm_stage0<<<dim3(1), dim3(256), 0, stream>>>(fc1, rpe, w2, b2, hops, wpack, bsum);
    hipMemsetAsync(gsum, 0, 2 * 64 * sizeof(float), stream);

    dim3 grid(GH, 128);                           // (head, batch) persistent blocks
    gm_main<<<grid, dim3(96), LDS_TOT, stream>>>(x, wpack, bsum, gsum, gsumsq,
                                                 scaleA, shiftA, out, 0);
    gm_finalize<<<dim3(1), dim3(64), 0, stream>>>(gsum, gsumsq, gamma, beta,
                                                  scaleA, shiftA);
    gm_main<<<grid, dim3(96), LDS_TOT, stream>>>(x, wpack, bsum, gsum, gsumsq,
                                                 scaleA, shiftA, out, 1);
}
